// rnn_27006754358056
// MI455X (gfx1250) — matrix operand order repaired and checked by reading, not compiled, not run on hardware
//
#include <hip/hip_runtime.h>

// ---------------- problem dims ----------------
#define B_DIM 128
#define T_DIM 512
#define E_DIM 1024
#define H_DIM 1024
#define V_DIM 1024

// ---------------- WMMA types ----------------
typedef __attribute__((ext_vector_type(16))) __bf16 v16bf;
typedef __attribute__((ext_vector_type(8)))  float  v8f;

union ABFrag { v16bf v; uint4 q[2]; };

// float -> bf16 round-to-nearest-even (bit manipulation; no __bf16 arithmetic)
__device__ __forceinline__ unsigned short f32_to_bf16(float f) {
  unsigned int u = __float_as_uint(f);
  unsigned int r = u + 0x7FFFu + ((u >> 16) & 1u);
  if ((u & 0x7F800000u) == 0x7F800000u) r = u;  // keep inf/nan payload
  return (unsigned short)(r >> 16);
}

// A fragment (16x32 bf16, M x K) from a row-major row pointer.
// lane L: m = L&15 (row), hi = L>>4. v0-3 = K[k0+hi*8 .. +7], v4-7 = K[k0+16+hi*8 .. +7]
__device__ __forceinline__ v16bf load_a_frag(const unsigned short* rowp, int hi, int k0) {
  ABFrag u;
  const unsigned short* p = rowp + k0 + hi * 8;
  u.q[0] = *(const uint4*)(p);
  u.q[1] = *(const uint4*)(p + 16);
  return u.v;
}

// B fragment (32x16 bf16, K x N) where B(k,n) = W[n][k], W row-major [N_total, ldk].
// lane L: n = L&15 (col), hi = L>>4. v0-3 = K[k0+hi*8 .. +7], v4-7 = K[k0+16+hi*8 .. +7]
__device__ __forceinline__ v16bf load_b_frag(const unsigned short* W, int ldk,
                                             int nGlobal, int hi, int k0) {
  ABFrag u;
  const unsigned short* p = W + (size_t)nGlobal * ldk + k0 + hi * 8;
  u.q[0] = *(const uint4*)(p);
  u.q[1] = *(const uint4*)(p + 16);
  return u.v;
}

#define WMMA_BF16(a, b, c) \
  __builtin_amdgcn_wmma_f32_16x16x32_bf16(false, (a), false, (b), (short)0, (c), \
                                          false, false)

// ---------------- device-wide barrier (all blocks resident) ----------------
__device__ __forceinline__ void grid_barrier(unsigned int* cnt, unsigned int* gen,
                                             unsigned int nBlocks) {
  __syncthreads();
  if (threadIdx.x == 0) {
    __threadfence();
    unsigned int g = __hip_atomic_load(gen, __ATOMIC_RELAXED, __HIP_MEMORY_SCOPE_AGENT);
    unsigned int arrived =
        __hip_atomic_fetch_add(cnt, 1u, __ATOMIC_ACQ_REL, __HIP_MEMORY_SCOPE_AGENT);
    if (arrived == nBlocks - 1u) {
      __hip_atomic_store(cnt, 0u, __ATOMIC_RELAXED, __HIP_MEMORY_SCOPE_AGENT);
      __hip_atomic_fetch_add(gen, 1u, __ATOMIC_RELEASE, __HIP_MEMORY_SCOPE_AGENT);
    } else {
      while (__hip_atomic_load(gen, __ATOMIC_ACQUIRE, __HIP_MEMORY_SCOPE_AGENT) == g) {
        __builtin_amdgcn_s_sleep(2);
      }
    }
  }
  __syncthreads();
}

// ---------------- small utility kernels ----------------
__global__ void cvt_f32_bf16_kernel(const float* __restrict__ src,
                                    unsigned short* __restrict__ dst, int n) {
  int i = blockIdx.x * blockDim.x + threadIdx.x;
  if (i < n) dst[i] = f32_to_bf16(src[i]);
}

__global__ void barrier_init_kernel(unsigned int* p) { p[0] = 0u; p[1] = 0u; }

// ---------------- phase 1: xp = embed[x] @ Wx^T + bx  (into d_out h-region) ----------
// 2x2 register blocking + explicit 1-deep software pipeline:
// iteration k's 4 WMMAs run while iteration k+1's 8 b128 loads are in flight.
__global__ void __launch_bounds__(256)
xproj_kernel(const int* __restrict__ x, const unsigned short* __restrict__ Ebf,
             const unsigned short* __restrict__ Wxbf, const float* __restrict__ bx,
             float* __restrict__ xp /* [B*T, H] */) {
  const int lane = threadIdx.x & 31;
  const int waveId = blockIdx.x * (blockDim.x >> 5) + (threadIdx.x >> 5);
  const int nColPairs = (H_DIM / 16) / 2;           // 32
  const int rowPair = waveId / nColPairs;           // 0..2047  (32 rows each)
  const int colPair = waveId % nColPairs;           // 0..31    (32 cols each)
  const int m  = lane & 15;
  const int hi = lane >> 4;

  const int rowBase = rowPair * 32;                 // (b*T + t) base
  const unsigned short* arow0 = Ebf + (size_t)x[rowBase + m] * E_DIM;
  const unsigned short* arow1 = Ebf + (size_t)x[rowBase + 16 + m] * E_DIM;
  const int n0 = colPair * 32 + m;                  // output feature, tile 0
  const int n1 = n0 + 16;                           // output feature, tile 1

  v8f c00 = {}, c01 = {}, c10 = {}, c11 = {};

  v16bf a0 = load_a_frag(arow0, hi, 0);
  v16bf a1 = load_a_frag(arow1, hi, 0);
  v16bf b0 = load_b_frag(Wxbf, E_DIM, n0, hi, 0);
  v16bf b1 = load_b_frag(Wxbf, E_DIM, n1, hi, 0);
#pragma unroll 4
  for (int k0 = 0; k0 < E_DIM - 32; k0 += 32) {
    v16bf a0n = load_a_frag(arow0, hi, k0 + 32);
    v16bf a1n = load_a_frag(arow1, hi, k0 + 32);
    v16bf b0n = load_b_frag(Wxbf, E_DIM, n0, hi, k0 + 32);
    v16bf b1n = load_b_frag(Wxbf, E_DIM, n1, hi, k0 + 32);
    c00 = WMMA_BF16(a0, b0, c00);
    c01 = WMMA_BF16(a0, b1, c01);
    c10 = WMMA_BF16(a1, b0, c10);
    c11 = WMMA_BF16(a1, b1, c11);
    a0 = a0n; a1 = a1n; b0 = b0n; b1 = b1n;
  }
  c00 = WMMA_BF16(a0, b0, c00);
  c01 = WMMA_BF16(a0, b1, c01);
  c10 = WMMA_BF16(a1, b0, c10);
  c11 = WMMA_BF16(a1, b1, c11);

  const float bias0 = bx[n0];
  const float bias1 = bx[n1];
#pragma unroll
  for (int r = 0; r < 8; ++r) {
    const size_t row0 = (size_t)(rowBase + r + 8 * hi);
    const size_t row1 = row0 + 16;
    xp[row0 * H_DIM + n0] = c00[r] + bias0;
    xp[row0 * H_DIM + n1] = c01[r] + bias1;
    xp[row1 * H_DIM + n0] = c10[r] + bias0;
    xp[row1 * H_DIM + n1] = c11[r] + bias1;
  }
}

// Pipelined 1x2 GEMM body for the scan: acc += Arow @ W^T for columns n0, n1.
__device__ __forceinline__ void gemm_1x2(const unsigned short* __restrict__ arow,
                                         const unsigned short* __restrict__ W,
                                         int n0, int n1, int hi,
                                         v8f& acc0, v8f& acc1) {
  v16bf a  = load_a_frag(arow, hi, 0);
  v16bf b0 = load_b_frag(W, H_DIM, n0, hi, 0);
  v16bf b1 = load_b_frag(W, H_DIM, n1, hi, 0);
#pragma unroll 4
  for (int k0 = 0; k0 < H_DIM - 32; k0 += 32) {
    v16bf an  = load_a_frag(arow, hi, k0 + 32);
    v16bf b0n = load_b_frag(W, H_DIM, n0, hi, k0 + 32);
    v16bf b1n = load_b_frag(W, H_DIM, n1, hi, k0 + 32);
    acc0 = WMMA_BF16(a, b0, acc0);
    acc1 = WMMA_BF16(a, b1, acc1);
    a = an; b0 = b0n; b1 = b1n;
  }
  acc0 = WMMA_BF16(a, b0, acc0);
  acc1 = WMMA_BF16(a, b1, acc1);
}

// ---------------- phase 2: persistent sequential scan ----------------
// grid = 32 blocks x 256 thr = 256 waves = 8 rowTiles(x16 batch) * 32 colPairs(x32 feat)
__global__ void __launch_bounds__(256)
scan_kernel(const float* __restrict__ bh, const float* __restrict__ bo,
            const unsigned short* __restrict__ Whbf,
            const unsigned short* __restrict__ Wobf,
            unsigned short* __restrict__ hbuf0, unsigned short* __restrict__ hbuf1,
            float* __restrict__ h_out /* [B,T,H]; pre-loaded with xp */,
            float* __restrict__ y_out /* [B,T,V] */,
            unsigned int* __restrict__ bar, unsigned int nBlocks) {
  const int lane = threadIdx.x & 31;
  const int waveId = blockIdx.x * (blockDim.x >> 5) + (threadIdx.x >> 5);
  const int nColPairs = (H_DIM / 16) / 2;           // 32
  const int rowTile = waveId / nColPairs;           // 0..7  (batch tile)
  const int colPair = waveId % nColPairs;           // 0..31
  const int m  = lane & 15;
  const int hi = lane >> 4;
  const int n0 = colPair * 32 + m;
  const int n1 = n0 + 16;
  const float bhv0 = bh[n0], bhv1 = bh[n1];
  const float bov0 = bo[n0], bov1 = bo[n1];
  const size_t arowOff = (size_t)(rowTile * 16 + m) * H_DIM;

  for (int t = 0; t < T_DIM; ++t) {
    const unsigned short* prev = (t & 1) ? hbuf1 : hbuf0;
    unsigned short*       cur  = (t & 1) ? hbuf0 : hbuf1;

    // ---- h_t = tanh(xp_t + bh + h_{t-1} @ Wh^T) ----
    v8f c0, c1;
#pragma unroll
    for (int r = 0; r < 8; ++r) {
      const size_t b = (size_t)(rowTile * 16 + r + 8 * hi);
      c0[r] = h_out[(b * T_DIM + t) * H_DIM + n0] + bhv0;
      c1[r] = h_out[(b * T_DIM + t) * H_DIM + n1] + bhv1;
    }
    gemm_1x2(prev + arowOff, Whbf, n0, n1, hi, c0, c1);
#pragma unroll
    for (int r = 0; r < 8; ++r) {
      const size_t b = (size_t)(rowTile * 16 + r + 8 * hi);
      const float hv0 = tanhf(c0[r]);
      const float hv1 = tanhf(c1[r]);
      h_out[(b * T_DIM + t) * H_DIM + n0] = hv0;       // overwrite xp with h
      h_out[(b * T_DIM + t) * H_DIM + n1] = hv1;
      cur[b * H_DIM + n0] = f32_to_bf16(hv0);          // bf16 copy for GEMMs
      cur[b * H_DIM + n1] = f32_to_bf16(hv1);
    }

    grid_barrier(bar, bar + 1, nBlocks);               // h_t visible everywhere

    // ---- y_t = h_t @ Wo^T + bo ----  (reads cur; next h-phase writes prev: no hazard)
    v8f cy0, cy1;
#pragma unroll
    for (int r = 0; r < 8; ++r) { cy0[r] = bov0; cy1[r] = bov1; }
    gemm_1x2(cur + arowOff, Wobf, n0, n1, hi, cy0, cy1);
#pragma unroll
    for (int r = 0; r < 8; ++r) {
      const size_t b = (size_t)(rowTile * 16 + r + 8 * hi);
      y_out[(b * T_DIM + t) * V_DIM + n0] = cy0[r];
      y_out[(b * T_DIM + t) * V_DIM + n1] = cy1[r];
    }
  }
}

// ---------------- host-side launch ----------------
extern "C" void kernel_launch(void* const* d_in, const int* in_sizes, int n_in,
                              void* d_out, int out_size, void* d_ws, size_t ws_size,
                              hipStream_t stream) {
  (void)in_sizes; (void)n_in; (void)out_size; (void)ws_size;

  const int*   x      = (const int*)d_in[0];
  const float* h_init = (const float*)d_in[1];
  const float* embed  = (const float*)d_in[2];
  const float* Wx     = (const float*)d_in[3];
  const float* bx     = (const float*)d_in[4];
  const float* Wh     = (const float*)d_in[5];
  const float* bh     = (const float*)d_in[6];
  const float* Wo     = (const float*)d_in[7];
  const float* bo     = (const float*)d_in[8];

  float* h_out = (float*)d_out;                          // [B,T,H]
  float* y_out = h_out + (size_t)B_DIM * T_DIM * H_DIM;  // [B,T,V]

  // workspace layout (~8.9 MB)
  unsigned int*   bar   = (unsigned int*)d_ws;
  unsigned short* Ebf   = (unsigned short*)((char*)d_ws + 256);
  unsigned short* Wxbf  = Ebf  + (size_t)V_DIM * E_DIM;
  unsigned short* Whbf  = Wxbf + (size_t)H_DIM * E_DIM;
  unsigned short* Wobf  = Whbf + (size_t)H_DIM * H_DIM;
  unsigned short* hbuf0 = Wobf + (size_t)V_DIM * H_DIM;
  unsigned short* hbuf1 = hbuf0 + (size_t)B_DIM * H_DIM;

  barrier_init_kernel<<<1, 1, 0, stream>>>(bar);

  const int NW = 1024 * 1024;  // elements per weight matrix / embed table
  cvt_f32_bf16_kernel<<<NW / 256, 256, 0, stream>>>(embed, Ebf, NW);
  cvt_f32_bf16_kernel<<<NW / 256, 256, 0, stream>>>(Wx, Wxbf, NW);
  cvt_f32_bf16_kernel<<<NW / 256, 256, 0, stream>>>(Wh, Whbf, NW);
  cvt_f32_bf16_kernel<<<NW / 256, 256, 0, stream>>>(Wo, Wobf, NW);
  cvt_f32_bf16_kernel<<<(B_DIM * H_DIM) / 256, 256, 0, stream>>>(h_init, hbuf0,
                                                                 B_DIM * H_DIM);

  // Phase 1: 2x2 tiles per wave: (B*T/32) * (H/32) waves = 65536 -> 8192 blocks
  {
    const int waves = (B_DIM * T_DIM / 32) * (H_DIM / 32);
    xproj_kernel<<<waves / 8, 256, 0, stream>>>(x, Ebf, Wxbf, bx, h_out);
  }

  // Phase 2: persistent scan, 32 blocks (256 waves) + device-wide barrier per step
  scan_kernel<<<32, 256, 0, stream>>>(bh, bo, Whbf, Wobf, hbuf0, hbuf1, h_out, y_out,
                                      bar, 32u);
}
